// RelationAttn_2954937500138
// MI455X (gfx1250) — matrix operand order repaired and checked by reading, not compiled, not run on hardware
//
#include <hip/hip_runtime.h>

#define BATCH 8
#define SEQ   2048
#define DIM   1024

typedef __attribute__((ext_vector_type(16))) __bf16 v16bf;
typedef __attribute__((ext_vector_type(8)))  __bf16 v8bf;
typedef __attribute__((ext_vector_type(8)))  float  v8f;

__device__ __forceinline__ v8f wmma_bf16(v16bf a, v16bf b, v8f c) {
    // D = A(16x32 bf16) * B(32x16 bf16) + C(16x16 f32)
    return __builtin_amdgcn_wmma_f32_16x16x32_bf16(
        /*neg_a=*/false, a, /*neg_b=*/false, b,
        /*c_mod=*/(short)0, c, /*reuse_a=*/false, /*reuse_b=*/false);
}

// A-matrix 16x32 bf16 fragment: lane holds row M=lane%16, K-half = lane/16.
// elements [0..15] -> K = base + kh*16 + i
// caller passes p = &A[row][base + kh*16]; one contiguous 16-half run (two 16B loads).
__device__ __forceinline__ v16bf load_afrag(const __bf16* p) {
    v8bf lo = *(const v8bf*)(p);
    v8bf hi = *(const v8bf*)(p + 8);
    v16bf a;
#pragma unroll
    for (int i = 0; i < 8; ++i) { a[i] = lo[i]; a[i + 8] = hi[i]; }
    return a;
}

// ---------- prep kernels ----------
__global__ void k_cvt_bf16(const float* __restrict__ src, __bf16* __restrict__ dst, int n) {
    int i = blockIdx.x * blockDim.x + threadIdx.x;
    if (i < n) dst[i] = (__bf16)src[i];
}

// EbT[b][d][n] = enc[b][n][d]  (key-contiguous copy for V fragments)
__global__ void k_trans_enc(const float* __restrict__ enc, __bf16* __restrict__ EbT) {
    int i = blockIdx.x * blockDim.x + threadIdx.x;      // over BATCH*DIM*SEQ
    int n = i % SEQ;
    int d = (i / SEQ) % DIM;
    int b = i / (SEQ * DIM);
    EbT[i] = (__bf16)enc[((size_t)(b * SEQ + n)) * DIM + d];
}

// Rt[n][k] = R[k][n]  (k-contiguous for GEMM B fragments)
__global__ void k_trans_R(const float* __restrict__ R, __bf16* __restrict__ Rt) {
    int i = blockIdx.x * blockDim.x + threadIdx.x;      // over DIM*DIM
    int k = i % DIM;
    int n = i / DIM;
    Rt[i] = (__bf16)R[(size_t)k * DIM + n];
}

// ---------- Q = enc @ R  (bf16 in, f32 acc, bf16 out) ----------
// one wave computes a 16x32 tile; block = 4 waves; grid = 1024 row-tiles * 8
__global__ __launch_bounds__(128) void k_qgemm(const __bf16* __restrict__ Eb,
                                               const __bf16* __restrict__ Rt,
                                               __bf16* __restrict__ Qb) {
    const int w = threadIdx.x >> 5, lane = threadIdx.x & 31;
    const int r = lane & 15, h = lane >> 4;
    const int mt = blockIdx.x >> 3;                     // row tile (16 rows)
    const int ct = ((blockIdx.x & 7) << 2) + w;         // col tile (32 cols)

    const __bf16* Abase = Eb + ((size_t)(mt * 16 + r)) * DIM + h * 16;
    const __bf16* B0    = Rt + ((size_t)(ct * 32 + r)) * DIM + h * 16;  // n = ct*32 + lane%16
    const __bf16* B1    = B0 + (size_t)16 * DIM;                        // n += 16

    v8f c0 = {}, c1 = {};
#pragma unroll 4
    for (int kk = 0; kk < DIM / 32; ++kk) {
        v16bf a  = load_afrag(Abase + kk * 32);
        v16bf b0 = *(const v16bf*)(B0 + kk * 32);       // K contiguous: 16 halves
        v16bf b1 = *(const v16bf*)(B1 + kk * 32);
        c0 = wmma_bf16(a, b0, c0);
        c1 = wmma_bf16(a, b1, c1);
    }
#pragma unroll
    for (int r0 = 0; r0 < 8; ++r0) {
        size_t row = (size_t)(mt * 16 + r0 + 8 * h);
        Qb[row * DIM + ct * 32 + r]      = (__bf16)c0[r0];
        Qb[row * DIM + ct * 32 + 16 + r] = (__bf16)c1[r0];
    }
}

// ---------- fused causal flash attention ----------
// block = 4 waves, owns 16 query rows. Waves split d=1024 4-way:
//  S partials via WMMA -> LDS -> summed; online softmax redundant per wave;
//  each wave accumulates O over its 256-dim slice with bf16 WMMA.
__global__ __launch_bounds__(128) void k_attn(const __bf16* __restrict__ Qb,
                                              const __bf16* __restrict__ Eb,
                                              const __bf16* __restrict__ EbT,
                                              float* __restrict__ out) {
    const int b  = blockIdx.x >> 7;        // /128 q-blocks
    const int qb = blockIdx.x & 127;
    const int w = threadIdx.x >> 5, lane = threadIdx.x & 31;
    const int r = lane & 15, h = lane >> 4;

    __shared__ float  Sp[4][16][32];
    __shared__ __bf16 Pt[16][32];

    // Q fragments for this wave's k-slice [w*256, w*256+256), kept in VGPRs.
    v16bf qf[8];
    const __bf16* Qbase = Qb + ((size_t)(b * SEQ + qb * 16 + r)) * DIM + w * 256 + h * 16;
#pragma unroll
    for (int kk = 0; kk < 8; ++kk) qf[kk] = load_afrag(Qbase + kk * 32);

    v8f o[16];
#pragma unroll
    for (int t = 0; t < 16; ++t) o[t] = (v8f){};
    float m_r = -1e30f, l_r = 0.0f;

    const int nkb = (qb * 16 + 15) / 32 + 1;            // causal: key blocks of 32
    for (int jb = 0; jb < nkb; ++jb) {
        const int keyBase = jb * 32;

        // ---- S partial over this wave's 256-dim slice ----
        v8f s0 = {}, s1 = {};
        const __bf16* K0 = Eb + ((size_t)(b * SEQ + keyBase + r)) * DIM + w * 256 + h * 16;
        const __bf16* K1 = K0 + (size_t)16 * DIM;
#pragma unroll
        for (int kk = 0; kk < 8; ++kk) {
            v16bf b0 = *(const v16bf*)(K0 + kk * 32);
            v16bf b1 = *(const v16bf*)(K1 + kk * 32);
            s0 = wmma_bf16(qf[kk], b0, s0);
            s1 = wmma_bf16(qf[kk], b1, s1);
        }
#pragma unroll
        for (int r0 = 0; r0 < 8; ++r0) {
            Sp[w][r0 + 8 * h][r]      = s0[r0];
            Sp[w][r0 + 8 * h][16 + r] = s1[r0];
        }
        __syncthreads();

        // ---- online softmax (each wave redundantly; lane pair handles row r) ----
        float sv[16];
        float mloc = -1e30f;
#pragma unroll
        for (int j = 0; j < 16; ++j) {
            int col = h * 16 + j;
            float s = Sp[0][r][col] + Sp[1][r][col] + Sp[2][r][col] + Sp[3][r][col];
            if (keyBase + col > qb * 16 + r) s = -1e30f;    // causal mask
            sv[j] = s;
            mloc = fmaxf(mloc, s);
        }
        float m2    = fmaxf(mloc, __shfl_xor(mloc, 16, 32));
        float m_new = fmaxf(m_r, m2);
        float scale = __expf(m_r - m_new);
        float lloc  = 0.0f;
#pragma unroll
        for (int j = 0; j < 16; ++j) {
            float p = __expf(sv[j] - m_new);
            sv[j] = p;
            lloc += p;
        }
        l_r = l_r * scale + lloc + __shfl_xor(lloc, 16, 32);
        m_r = m_new;
        if (w == 0) {
#pragma unroll
            for (int j = 0; j < 16; ++j) Pt[r][h * 16 + j] = (__bf16)sv[j];
        }
        __syncthreads();

        // ---- rescale O and accumulate O += P * V over this wave's dim slice ----
        float sc[8];
#pragma unroll
        for (int r0 = 0; r0 < 8; ++r0) sc[r0] = __shfl(scale, r0 + 8 * h, 32);
#pragma unroll
        for (int t = 0; t < 16; ++t)
#pragma unroll
            for (int r0 = 0; r0 < 8; ++r0) o[t][r0] *= sc[r0];

        v16bf pf = load_afrag(&Pt[r][h * 16]);
#pragma unroll
        for (int t = 0; t < 16; ++t) {
            const __bf16* vb = EbT + ((size_t)(b * DIM + w * 256 + t * 16 + r)) * SEQ
                                   + keyBase + h * 16;  // key-contiguous 16 halves
            v16bf vf = *(const v16bf*)(vb);
            o[t] = wmma_bf16(pf, vf, o[t]);
        }
        // next iteration's Sp write is fenced by its own post-write barrier;
        // Pt reads above are fenced by the barrier after the next Sp write.
    }

    // ---- finalize: divide by l and store f32 ----
    float li[8];
#pragma unroll
    for (int r0 = 0; r0 < 8; ++r0) li[r0] = __shfl(l_r, r0 + 8 * h, 32);
#pragma unroll
    for (int t = 0; t < 16; ++t) {
#pragma unroll
        for (int r0 = 0; r0 < 8; ++r0) {
            size_t row = (size_t)(b * SEQ + qb * 16 + r0 + 8 * h);
            out[row * DIM + w * 256 + t * 16 + r] = o[t][r0] / li[r0];
        }
    }
}

extern "C" void kernel_launch(void* const* d_in, const int* in_sizes, int n_in,
                              void* d_out, int out_size, void* d_ws, size_t ws_size,
                              hipStream_t stream) {
    const float* enc = (const float*)d_in[0];   // [8,2048,1024] f32
    const float* R   = (const float*)d_in[1];   // [1024,1024] f32
    float* out = (float*)d_out;                 // [8,2048,1024] f32

    const size_t nE = (size_t)BATCH * SEQ * DIM;        // 16,777,216
    __bf16* Eb  = (__bf16*)d_ws;                        // [B*SEQ][DIM]     32 MB
    __bf16* EbT = Eb  + nE;                             // [B][DIM][SEQ]    32 MB
    __bf16* Rt  = EbT + nE;                             // [DIM][DIM] (R^T)  2 MB
    __bf16* Qb  = Rt  + (size_t)DIM * DIM;              // [B*SEQ][DIM]     32 MB

    k_cvt_bf16 <<<(int)(nE / 256), 256, 0, stream>>>(enc, Eb, (int)nE);
    k_trans_enc<<<(int)(nE / 256), 256, 0, stream>>>(enc, EbT);
    k_trans_R  <<<DIM * DIM / 256, 256, 0, stream>>>(R, Rt);

    k_qgemm<<<(BATCH * SEQ / 16) * 8, 128, 0, stream>>>(Eb, Rt, Qb);   // 8192 blocks
    k_attn <<<BATCH * (SEQ / 16),    128, 0, stream>>>(Qb, Eb, EbT, out); // 1024 blocks
}
